// InfoGatAttn_79559974191128
// MI455X (gfx1250) — matrix operand order repaired and checked by reading, not compiled, not run on hardware
//
#include <hip/hip_runtime.h>

// Problem constants (from reference setup_inputs)
#define N_SEQ   64
#define D_MODEL 256
#define N_HEAD  8
#define HH      32
#define XS      272   // padded bf16 row stride for X in LDS (544B, multiple of 32B)
#define PS      260   // padded f32 row stride for proj in LDS

typedef __attribute__((ext_vector_type(16))) __bf16 v16bf;
typedef __attribute__((ext_vector_type(8)))  __bf16 v8bf;
typedef __attribute__((ext_vector_type(8)))  float  v8f;

__device__ __forceinline__ unsigned short f32_to_bf16(float x) {
    unsigned int u = __float_as_uint(x);
    unsigned int r = u + 0x7FFFu + ((u >> 16) & 1u);   // round-to-nearest-even
    return (unsigned short)(r >> 16);
}

// ---------------- prep: W (f32 DxD) -> bf16 in workspace ----------------
__global__ void wconv_kernel(const float* __restrict__ W,
                             unsigned short* __restrict__ Wbf, int n) {
    int i = blockIdx.x * blockDim.x + threadIdx.x;
    if (i < n) Wbf[i] = f32_to_bf16(W[i]);
}

// ---------------- main fused kernel: one workgroup per batch ----------------
__global__ __launch_bounds__(256)
void infogat_kernel(const float* __restrict__ X,          // [B,N,D] f32
                    const int*   __restrict__ lengths,    // [B]
                    const unsigned short* __restrict__ Wbf, // [D,D] bf16
                    const float* __restrict__ cur_a,      // [H,hh]
                    const float* __restrict__ nb_a,       // [H,hh]
                    const float* __restrict__ Wout,       // [D,D] f32
                    float* __restrict__ out,              // [B*D attn | B*D query]
                    int Btot)
{
    // X-bf16 staging (64*272*2 = 34816B) aliased with proj f32 (64*260*4 = 66560B)
    __shared__ __align__(32) unsigned char s_buf[N_SEQ * PS * 4];
    __shared__ float s_cur[N_HEAD];
    __shared__ float s_score[N_SEQ * N_HEAD];
    __shared__ float s_attn[N_SEQ * N_HEAD];
    __shared__ __align__(16) float s_ai[D_MODEL];

    const int b    = blockIdx.x;
    const int t    = threadIdx.x;
    const int lane = t & 31;
    const int wave = t >> 5;

    unsigned short* Xbf = (unsigned short*)s_buf;
    float*          proj = (float*)s_buf;

    const float* Xb = X + (size_t)b * (N_SEQ * D_MODEL);

    // ---- Phase 0: stage X[b] into LDS as bf16 (coalesced) ----
    for (int r = 0; r < N_SEQ; ++r)
        Xbf[r * XS + t] = f32_to_bf16(Xb[r * D_MODEL + t]);
    __syncthreads();

    // ---- Phase 1: proj[64x256] = Xb @ W^T via v_wmma_f32_16x16x32_bf16 ----
    // wave -> (mtile 0..3, 8 etiles); 64 WMMA per wave (8 ksteps x 8 etiles)
    const int mtile = wave & 3;
    const int ebase = (wave >> 2) * 8;
    const int half  = lane >> 4;          // 0: lanes 0-15, 1: lanes 16-31
    const int l15   = lane & 15;

    v8f c[8] = {};

    const int arow = mtile * 16 + l15;
#pragma unroll
    for (int ks = 0; ks < 8; ++ks) {
        // A (16x32 bf16): lane holds contiguous K-chunk [half*16..+16) (matches B)
        const unsigned short* ap = Xbf + arow * XS + ks * 32 + half * 16;
        const v8bf a_lo = *(const v8bf*)(ap);
        const v8bf a_hi = *(const v8bf*)(ap + 8);
        const v16bf a = __builtin_shufflevector(a_lo, a_hi,
                            0,1,2,3,4,5,6,7,8,9,10,11,12,13,14,15);
#pragma unroll
        for (int et = 0; et < 8; ++et) {
            // B (32x16 bf16): column e = W row e; lane holds contiguous K-chunk of 16
            const int e0 = (ebase + et) * 16 + l15;
            const v16bf bb = *(const v16bf*)(Wbf + (size_t)e0 * D_MODEL + ks * 32 + half * 16);
            c[et] = __builtin_amdgcn_wmma_f32_16x16x32_bf16(
                        false, a, false, bb, (short)0, c[et], false, false);
        }
    }
    __syncthreads();   // all X reads done before aliasing s_buf as proj

    // scatter C fragments (f32 16x16 layout: VGPR v -> M=v (+8 for hi lanes), N=lane&15)
    const int prow0 = mtile * 16 + half * 8;
#pragma unroll
    for (int et = 0; et < 8; ++et) {
        const int pcol = (ebase + et) * 16 + l15;
#pragma unroll
        for (int v = 0; v < 8; ++v)
            proj[(prow0 + v) * PS + pcol] = c[et][v];
    }
    __syncthreads();

    // ---- Phase 2: attention (cheap VALU work) ----
    int len = lengths[b]; if (len < 1) len = 1;

    if (t < N_HEAD) {
        float s = 0.f;
#pragma unroll
        for (int j = 0; j < HH; ++j) s += proj[t * HH + j] * cur_a[t * HH + j];
        s_cur[t] = s;
    }
#pragma unroll
    for (int rep = 0; rep < 2; ++rep) {
        const int p = t + rep * 256;
        const int h = p & 7, n = p >> 3;
        float s = 0.f;
#pragma unroll
        for (int j = 0; j < HH; ++j) s += proj[n * PS + h * HH + j] * nb_a[h * HH + j];
        s_score[n * 8 + h] = s;
    }
    __syncthreads();

#pragma unroll
    for (int rep = 0; rep < 2; ++rep) {
        const int p = t + rep * 256;
        const int h = p & 7, n = p >> 3;
        float sc = s_cur[h] + s_score[n * 8 + h];
        sc = sc > 0.f ? sc : 0.01f * sc;                 // leaky_relu
        s_score[n * 8 + h] = sc;
    }
    __syncthreads();

    // per-head masked softmax over n (threads 0..7)
    if (t < N_HEAD) {
        float m = -3.4e38f;
        for (int n = 0; n < len; ++n) m = fmaxf(m, s_score[n * 8 + t]);
        float sum = 0.f;
        for (int n = 0; n < len; ++n) sum += __expf(s_score[n * 8 + t] - m);
        const float inv = 1.f / sum;
        for (int n = 0; n < N_SEQ; ++n)
            s_attn[n * 8 + t] = (n < len) ? __expf(s_score[n * 8 + t] - m) * inv : 0.f;
    }
    __syncthreads();

    // attn_info[d] = sum_n attn[n,h] * proj[n,d]   (thread t = d)
    {
        const int h = t >> 5;
        float acc = 0.f;
        for (int n = 0; n < len; ++n) acc += s_attn[n * 8 + h] * proj[n * PS + t];
        s_ai[t] = acc;
    }
    __syncthreads();

    // out[b,e] = attn_info . Wout[e,:]  (thread t = e), plus query-info copy
    {
        const float4* wr  = (const float4*)(Wout + (size_t)t * D_MODEL);
        const float4* ai4 = (const float4*)s_ai;
        float acc = 0.f;
#pragma unroll 4
        for (int q = 0; q < D_MODEL / 4; ++q) {
            const float4 w = wr[q]; const float4 a4 = ai4[q];
            acc += w.x * a4.x + w.y * a4.y + w.z * a4.z + w.w * a4.w;
        }
        out[(size_t)b * D_MODEL + t] = acc;
        out[(size_t)Btot * D_MODEL + (size_t)b * D_MODEL + t] = Xb[t];  // X[b,0,:]
    }
}

extern "C" void kernel_launch(void* const* d_in, const int* in_sizes, int n_in,
                              void* d_out, int out_size, void* d_ws, size_t ws_size,
                              hipStream_t stream) {
    const float* inter   = (const float*)d_in[0];
    const int*   lengths = (const int*)d_in[1];
    const float* W       = (const float*)d_in[2];
    const float* cur_a   = (const float*)d_in[3];
    const float* nb_a    = (const float*)d_in[4];
    const float* Wout    = (const float*)d_in[5];
    float* out = (float*)d_out;
    unsigned short* Wbf = (unsigned short*)d_ws;   // needs 128KB; ws is scratch
    const int Btot = in_sizes[1];                  // lengths count == B

    wconv_kernel<<<(D_MODEL * D_MODEL + 255) / 256, 256, 0, stream>>>(W, Wbf, D_MODEL * D_MODEL);
    infogat_kernel<<<Btot, 256, 0, stream>>>(inter, lengths, Wbf, cur_a, nb_a, Wout, out, Btot);
}
